// GCN_62715112456685
// MI455X (gfx1250) — matrix operand order repaired and checked by reading, not compiled, not run on hardware
//
#include <hip/hip_runtime.h>
#include <hip/hip_bf16.h>

// ---------------------------------------------------------------------------
// GCN forward for MI455X (gfx1250, wave32).
//   x:[N,1]  edge_index:[2,E]  batch:[N]
//   L1: 1->128 (outer product), L2: 128->64 (WMMA f16), L3: 64->32 (WMMA f16)
//   mean-pool per graph (64) -> FC 32->10
// Memory-bound: ~86MB of intermediates fits in the 192MB L2; scatter atomics
// are L2-resident. GEMMs use v_wmma_f32_16x16x32_f16 with pre-packed f16
// weight fragments (2x global_load_b128 per fragment) and 4 M-tiles per wave
// so B fragments are reused from registers.
// ---------------------------------------------------------------------------

typedef __attribute__((ext_vector_type(16))) _Float16 v16h;
typedef __attribute__((ext_vector_type(8)))  _Float16 v8h;
typedef __attribute__((ext_vector_type(8)))  float    v8f;
typedef __attribute__((ext_vector_type(4)))  float    v4f;
typedef __attribute__((ext_vector_type(2)))  float    v2f;

#define THREADS 256

// ---------------------------- small helpers --------------------------------

__global__ void zero_f32(float* __restrict__ p, int n) {
    int i = blockIdx.x * blockDim.x + threadIdx.x;
    if (i < n) p[i] = 0.0f;
}

__global__ void degree_kernel(const int* __restrict__ dst, float* __restrict__ deg, int E) {
    int e = blockIdx.x * blockDim.x + threadIdx.x;
    if (e < E) atomicAdd(&deg[dst[e]], 1.0f);
}

// in-place: deg -> rsqrt(deg + 1)   (+1 = self loop)
__global__ void dinv_kernel(float* __restrict__ deg, int n) {
    int i = blockIdx.x * blockDim.x + threadIdx.x;
    if (i < n) deg[i] = rsqrtf(deg[i] + 1.0f);
}

__global__ void norm_kernel(const int* __restrict__ src, const int* __restrict__ dst,
                            const float* __restrict__ dinv, float* __restrict__ norm, int E) {
    int e = blockIdx.x * blockDim.x + threadIdx.x;
    if (e < E) norm[e] = dinv[src[e]] * dinv[dst[e]];
}

// layer-1 "GEMM": F_in == 1, so xw[i][f] = x[i] * W1[f]
__global__ void l1_outer(const float* __restrict__ x, const float* __restrict__ W1,
                         float* __restrict__ xw, int total) {
    int idx = blockIdx.x * blockDim.x + threadIdx.x;
    if (idx >= total) return;
    int i = idx >> 7;        // / 128
    int f = idx & 127;
    xw[idx] = x[i] * W1[f];
}

// ------------------------ edge scatter aggregation -------------------------
// One wave per edge; lanes cover the F features with b128/b64 gathers.
// f32 atomics land in the 192MB L2.
template <int F>
__global__ void edge_agg(const float* __restrict__ xw, const float* __restrict__ norm,
                         const int* __restrict__ src, const int* __restrict__ dst,
                         float* __restrict__ agg, int E) {
    constexpr int V = F / 32;
    int wavesPerBlock = blockDim.x >> 5;
    int e = blockIdx.x * wavesPerBlock + (threadIdx.x >> 5);
    if (e >= E) return;
    int lane = threadIdx.x & 31;

    int   s = src[e];
    int   d = dst[e];
    float w = norm[e];

    // hide gather latency: prefetch next edge's source row (global_prefetch_b8)
    if (e + 1 < E)
        __builtin_prefetch(xw + (size_t)src[e + 1] * F + lane * V, 0, 0);

    const float* xs = xw  + (size_t)s * F + lane * V;
    float*       ad = agg + (size_t)d * F + lane * V;
    if constexpr (V == 4) {
        v4f xv = *(const v4f*)xs;
#pragma unroll
        for (int j = 0; j < 4; ++j) atomicAdd(&ad[j], xv[j] * w);
    } else if constexpr (V == 2) {
        v2f xv = *(const v2f*)xs;
#pragma unroll
        for (int j = 0; j < 2; ++j) atomicAdd(&ad[j], xv[j] * w);
    } else {
        atomicAdd(&ad[0], xs[0] * w);
    }
}

// self-loop term + bias + ReLU; optionally emit f16 copy for next WMMA GEMM
template <int F, bool CVT>
__global__ void self_bias_relu(const float* __restrict__ xw, const float* __restrict__ dinv,
                               const float* __restrict__ b, float* __restrict__ agg,
                               _Float16* __restrict__ h16, int total) {
    int idx = blockIdx.x * blockDim.x + threadIdx.x;
    if (idx >= total) return;
    int i = idx / F;
    int f = idx % F;
    float di = dinv[i];
    float v  = agg[idx] + xw[idx] * di * di + b[f];
    v = fmaxf(v, 0.0f);
    agg[idx] = v;
    if (CVT) h16[idx] = (_Float16)v;
}

// -------------------- weight fragment pre-pack (f16) -----------------------
// Wfrag linear layout: (((nT * (K/32) + kc) * 32 + lane) * 16 + e)
// holding W[kc*32 + (lane>>4)*16 + e][nT*16 + (lane&15)], i.e. the exact
// per-lane v16h B operand in this file's contiguous k order (k = 16*(lane>>4) + e), same as A.
template <int K, int NOUT>
__global__ void pack_wfrag(const float* __restrict__ W, _Float16* __restrict__ Wfrag) {
    int idx = blockIdx.x * blockDim.x + threadIdx.x;
    if (idx >= K * NOUT) return;
    constexpr int KC = K / 32;
    int e    = idx & 15;
    int lane = (idx >> 4) & 31;
    int rest = idx >> 9;            // == nT*KC + kc
    int kc   = rest % KC;
    int nT   = rest / KC;
    int k = kc * 32 + (lane >> 4) * 16 + e;
    int n = nT * 16 + (lane & 15);
    Wfrag[idx] = (_Float16)W[(size_t)k * NOUT + n];
}

// ------------------------------ WMMA GEMM ----------------------------------
// D[M,NOUT] = A[M,K](f16 row-major) * W[K,NOUT] (pre-packed fragments).
// One wave handles MT consecutive 16-row tiles for one 16-col tile, keeping
// all K/32 B fragments in registers. Fully unrolled -> MT*(K/32) v_wmma ops.
// All guards are wave-uniform, so EXEC is all-ones at every WMMA.
template <int K, int NOUT, int MT>
__global__ void wmma_gemm_f16(const _Float16* __restrict__ A, const _Float16* __restrict__ Wfrag,
                              float* __restrict__ D, int M) {
    constexpr int KC     = K / 32;
    constexpr int nTiles = NOUT >> 4;
    int lane  = threadIdx.x & 31;
    int wave  = threadIdx.x >> 5;
    int group = blockIdx.x * (blockDim.x >> 5) + wave;
    int mGroups = M / (16 * MT);
    int nT = group / mGroups;
    int mG = group % mGroups;
    if (nT >= nTiles) return;       // exact grid; wave-uniform anyway

    int laneLo = lane & 15;
    int laneHi = lane >> 4;

    // ---- B fragments for this column tile: 2x b128 each, register-resident
    v16h b[KC];
    const _Float16* fb = Wfrag + (size_t)nT * KC * 512 + lane * 16;
#pragma unroll
    for (int kc = 0; kc < KC; ++kc) {
        v8h lo = *(const v8h*)(fb + kc * 512);
        v8h hi = *(const v8h*)(fb + kc * 512 + 8);
        v16h t;
#pragma unroll
        for (int j = 0; j < 8; ++j) { t[j] = lo[j]; t[8 + j] = hi[j]; }
        b[kc] = t;
    }

#pragma unroll
    for (int mi = 0; mi < MT; ++mi) {
        int mBase = (mG * MT + mi) * 16;
        const _Float16* arow = A + (size_t)(mBase + laneLo) * K;
        v8f acc = {};
#pragma unroll
        for (int kc = 0; kc < KC; ++kc) {
            int ka = kc * 32 + laneHi * 16;
            v8h alo = *(const v8h*)(arow + ka);
            v8h ahi = *(const v8h*)(arow + ka + 8);
            v16h a;
#pragma unroll
            for (int j = 0; j < 8; ++j) { a[j] = alo[j]; a[8 + j] = ahi[j]; }
            acc = __builtin_amdgcn_wmma_f32_16x16x32_f16(
                /*neg_a=*/false, a, /*neg_b=*/false, b[kc],
                /*c_mod=*/(short)0, acc, /*reuse_a=*/false, /*reuse_b=*/false);
        }
        float* dptr = D + (size_t)(mBase + laneHi * 8) * NOUT + nT * 16 + laneLo;
#pragma unroll
        for (int r = 0; r < 8; ++r)
            dptr[(size_t)r * NOUT] = acc[r];
    }
}

// ------------------------------ pool + FC ----------------------------------

__global__ void pool_kernel(const float* __restrict__ h, const int* __restrict__ batch,
                            float* __restrict__ pool, float* __restrict__ cnt, int n) {
    int idx = blockIdx.x * blockDim.x + threadIdx.x;
    if (idx >= n * 32) return;
    int i = idx >> 5;
    int f = idx & 31;
    int g = batch[i];
    atomicAdd(&pool[g * 32 + f], h[idx]);
    if (f == 0) atomicAdd(&cnt[g], 1.0f);
}

__global__ void fc_kernel(const float* __restrict__ pool, const float* __restrict__ cnt,
                          const float* __restrict__ Wfc, const float* __restrict__ bfc,
                          float* __restrict__ out) {
    int idx = blockIdx.x * blockDim.x + threadIdx.x;
    if (idx >= 64 * 10) return;
    int g = idx / 10;
    int j = idx % 10;
    float inv = 1.0f / fmaxf(cnt[g], 1.0f);
    float acc = bfc[j];
#pragma unroll
    for (int f = 0; f < 32; ++f)
        acc += pool[g * 32 + f] * inv * Wfc[f * 10 + j];
    out[idx] = acc;
}

// ------------------------------- launcher ----------------------------------

extern "C" void kernel_launch(void* const* d_in, const int* in_sizes, int n_in,
                              void* d_out, int out_size, void* d_ws, size_t ws_size,
                              hipStream_t stream) {
    const float* x    = (const float*)d_in[0];
    const int*   edge = (const int*)d_in[1];
    const int*   batch= (const int*)d_in[2];
    const float* W1   = (const float*)d_in[3];
    const float* b1   = (const float*)d_in[4];
    const float* W2   = (const float*)d_in[5];
    const float* b2   = (const float*)d_in[6];
    const float* W3   = (const float*)d_in[7];
    const float* b3   = (const float*)d_in[8];
    const float* Wfc  = (const float*)d_in[9];
    const float* bfc  = (const float*)d_in[10];
    float*       out  = (float*)d_out;

    const int N = in_sizes[0];        // 65536 (x is [N,1])
    const int E = in_sizes[1] / 2;    // 524288
    const int* src = edge;
    const int* dst = edge + E;

    // ---- workspace layout ----
    char* ws = (char*)d_ws;
    size_t off = 0;
    float* dinv = (float*)(ws + off); off += (size_t)N * 4;
    float* norm = (float*)(ws + off); off += (size_t)E * 4;
    float* xw   = (float*)(ws + off); off += (size_t)N * 128 * 4;
    float* agg  = (float*)(ws + off); off += (size_t)N * 128 * 4;
    _Float16* h16    = (_Float16*)(ws + off); off += (size_t)N * 128 * 2;
    float* pool = (float*)(ws + off); off += 64 * 32 * 4;
    float* cnt  = (float*)(ws + off); off += 64 * 4;
    _Float16* wfrag2 = (_Float16*)(ws + off); off += 128 * 64 * 2;
    _Float16* wfrag3 = (_Float16*)(ws + off); off += 64 * 32 * 2;

    auto gridOf = [](long n) { return (int)((n + THREADS - 1) / THREADS); };

    // ---- weight fragment packing (tiny, overlaps nothing it needs) ----
    pack_wfrag<128, 64><<<gridOf(128 * 64), THREADS, 0, stream>>>(W2, wfrag2);
    pack_wfrag<64, 32><<<gridOf(64 * 32), THREADS, 0, stream>>>(W3, wfrag3);

    // ---- normalization terms ----
    zero_f32<<<gridOf(N), THREADS, 0, stream>>>(dinv, N);
    degree_kernel<<<gridOf(E), THREADS, 0, stream>>>(dst, dinv, E);
    dinv_kernel<<<gridOf(N), THREADS, 0, stream>>>(dinv, N);
    norm_kernel<<<gridOf(E), THREADS, 0, stream>>>(src, dst, dinv, norm, E);

    const int edgeBlocks = (E + 7) / 8;   // 8 waves / block, 1 wave / edge

    // ---- layer 1: 1 -> 128 ----
    l1_outer<<<gridOf((long)N * 128), THREADS, 0, stream>>>(x, W1, xw, N * 128);
    zero_f32<<<gridOf((long)N * 128), THREADS, 0, stream>>>(agg, N * 128);
    edge_agg<128><<<edgeBlocks, THREADS, 0, stream>>>(xw, norm, src, dst, agg, E);
    self_bias_relu<128, true><<<gridOf((long)N * 128), THREADS, 0, stream>>>(
        xw, dinv, b1, agg, h16, N * 128);

    // ---- layer 2: 128 -> 64 (WMMA, MT=4 m-tiles per wave) ----
    {
        int groups = (64 / 16) * (N / 64);          // nTiles * mGroups
        wmma_gemm_f16<128, 64, 4><<<groups / 8, THREADS, 0, stream>>>(h16, wfrag2, xw, N);
    }
    zero_f32<<<gridOf((long)N * 64), THREADS, 0, stream>>>(agg, N * 64);
    edge_agg<64><<<edgeBlocks, THREADS, 0, stream>>>(xw, norm, src, dst, agg, E);
    self_bias_relu<64, true><<<gridOf((long)N * 64), THREADS, 0, stream>>>(
        xw, dinv, b2, agg, h16, N * 64);

    // ---- layer 3: 64 -> 32 (WMMA, MT=4) ----
    {
        int groups = (32 / 16) * (N / 64);
        wmma_gemm_f16<64, 32, 4><<<groups / 8, THREADS, 0, stream>>>(h16, wfrag3, xw, N);
    }
    zero_f32<<<gridOf((long)N * 32), THREADS, 0, stream>>>(agg, N * 32);
    edge_agg<32><<<edgeBlocks, THREADS, 0, stream>>>(xw, norm, src, dst, agg, E);
    self_bias_relu<32, false><<<gridOf((long)N * 32), THREADS, 0, stream>>>(
        xw, dinv, b3, agg, h16, N * 32);

    // ---- mean pool + FC ----
    zero_f32<<<gridOf(64 * 32 + 64), THREADS, 0, stream>>>(pool, 64 * 32 + 64); // pool+cnt contiguous
    pool_kernel<<<gridOf((long)N * 32), THREADS, 0, stream>>>(agg, batch, pool, cnt, N);
    fc_kernel<<<gridOf(640), THREADS, 0, stream>>>(pool, cnt, Wfc, bfc, out);
}
